// GraphAttention_72808285602179
// MI455X (gfx1250) — matrix operand order repaired and checked by reading, not compiled, not run on hardware
//
#include <hip/hip_runtime.h>
#include <cstdint>
#include <cstddef>

// ---------------------------------------------------------------------------
// Equivariant graph attention layer for MI455X (gfx1250, wave32).
// Edge-level GEMMs run on v_wmma_f32_16x16x32_f16: one wave per 16-edge
// M-tile, B (weight) fragments register-resident, A streamed from global,
// compile-time strides so addressing folds to immediate offsets.
// Irregular graph parts use VALU + atomics.
// ---------------------------------------------------------------------------

#define CSC   64     // scalar channels
#define CVC   32     // vector channels
#define HH    4      // heads
#define ASD   16     // attn scalar dim
#define AVD   8      // attn vector dim
#define NBF   16     // rbf basis
#define NODEW 160    // CS+3*CV

#define SQRT3_F 1.7320508075688772f
#define SQRT2_F 1.4142135623730951f
#define EPS_F   1e-5f

typedef __attribute__((ext_vector_type(16))) _Float16 v16h;
typedef __attribute__((ext_vector_type(8)))  _Float16 v8h;
typedef __attribute__((ext_vector_type(8)))  float    v8f;

// ------------------------------- helpers -----------------------------------

__device__ __forceinline__ float sigmoidf_(float x) { return 1.0f / (1.0f + expf(-x)); }

// order-preserving float<->uint encoding for atomicMax on signed floats
__device__ __forceinline__ unsigned encOrd(float f) {
    unsigned u = __float_as_uint(f);
    return (u & 0x80000000u) ? ~u : (u | 0x80000000u);
}
__device__ __forceinline__ float decOrd(unsigned u) {
    unsigned b = (u & 0x80000000u) ? (u ^ 0x80000000u) : ~u;
    return __uint_as_float(b);
}
#define ENC_NEG_INF 0x007FFFFFu   // encOrd(-inf)

// B fragment: 32x16 f16 tile from LDS-resident row-major B (ldb = N).
__device__ __forceinline__ v16h load_b_frag(const _Float16* __restrict__ Blds,
                                            int ldb, int k0, int col0, int lane) {
    int c = lane & 15, kh = lane >> 4;
    const _Float16* p = Blds + (size_t)(k0 + kh * 8) * ldb + col0 + c;
    v16h out;
#pragma unroll
    for (int j = 0; j < 16; ++j) out[j] = p[((j & 7) + (j >> 3) * 16) * ldb];
    return out;
}

// ------------------------- generic WMMA GEMM -------------------------------
// C[M x (NT*16)] (f16) = A[M x KDIM] (f16) * B[KDIM x NT*16] (f16), f32 acc.
// B fragments are gathered once per wave (via LDS) and stay in registers:
// NT*KDIM/32 fragments * 8 VGPRs (<= 96 VGPRs for the shapes used here).
// LDA/LDC are compile-time so per-row addressing folds to immediates.
// One wave per 16-row M tile, grid-stride over tiles. M multiple of 16.
template <int KDIM, int NT, int LDA, int LDC>
__global__ __launch_bounds__(256)
void wmma_gemm_kernel(const _Float16* __restrict__ A,
                      const _Float16* __restrict__ B,
                      _Float16* __restrict__ C, int mtiles) {
    constexpr int NDIM = NT * 16;
    constexpr int KS   = KDIM / 32;
    __shared__ _Float16 Blds[KDIM * NDIM];
    {
        const unsigned* src = (const unsigned*)B;
        unsigned* dst = (unsigned*)Blds;
        for (int i = threadIdx.x; i < KDIM * NDIM / 2; i += blockDim.x) dst[i] = src[i];
    }
    __syncthreads();

    const int lane = threadIdx.x & 31;
    const int wave = threadIdx.x >> 5;
    const int nw   = blockDim.x >> 5;
    const int r    = lane & 15;       // A row within tile / C column
    const int sel  = lane >> 4;       // A K-half select / C row-half select

    // one-time gather of all B fragments into registers
    v16h bfr[NT * KS];
#pragma unroll
    for (int nt = 0; nt < NT; ++nt)
#pragma unroll
        for (int ks = 0; ks < KS; ++ks)
            bfr[nt * KS + ks] = load_b_frag(Blds, NDIM, ks * 32, nt * 16, lane);

    for (int tile = blockIdx.x * nw + wave; tile < mtiles; tile += gridDim.x * nw) {
        // A fragments: lane holds row r, K-halves selected by sel.
        const _Float16* ap = A + (size_t)(tile * 16 + r) * LDA + sel * 8;
        v16h a[KS];
#pragma unroll
        for (int ks = 0; ks < KS; ++ks) {
            v8h lo = *(const v8h*)(ap + ks * 32);        // K = ks*32 + sel*8 + 0..7
            v8h hi = *(const v8h*)(ap + ks * 32 + 16);   // K = ks*32 + 16 + sel*8 + 0..7
#pragma unroll
            for (int i = 0; i < 8; ++i) { a[ks][i] = lo[i]; a[ks][i + 8] = hi[i]; }
        }

        // C tile base: lane writes column r, rows sel*8 + 0..7.
        _Float16* cp = C + (size_t)(tile * 16 + sel * 8) * LDC + r;
#pragma unroll
        for (int nt = 0; nt < NT; ++nt) {
            v8f acc = {0.f, 0.f, 0.f, 0.f, 0.f, 0.f, 0.f, 0.f};
#pragma unroll
            for (int ks = 0; ks < KS; ++ks)
                acc = __builtin_amdgcn_wmma_f32_16x16x32_f16(
                          false, a[ks], false, bfr[nt * KS + ks], (short)0, acc,
                          false, false);
#pragma unroll
            for (int i = 0; i < 8; ++i)
                cp[i * LDC + nt * 16] = (_Float16)acc[i];
        }
    }
}

// --------------------------- weight packing --------------------------------
__global__ void pack_f16_kernel(const float* __restrict__ src,
                                _Float16* __restrict__ dst, int n) {
    int i = blockIdx.x * blockDim.x + threadIdx.x;
    if (i < n) dst[i] = (_Float16)src[i];
}

// ------------------------------ init ---------------------------------------
__global__ void init_kernel(unsigned* amaxU, float* den, float* agg,
                            int n4, size_t nagg) {
    size_t i = (size_t)blockIdx.x * blockDim.x + threadIdx.x;
    size_t stride = (size_t)gridDim.x * blockDim.x;
    for (size_t j = i; j < nagg; j += stride) agg[j] = 0.f;
    for (size_t j = i; j < (size_t)n4; j += stride) { amaxU[j] = ENC_NEG_INF; den[j] = 0.f; }
}

// ------------------- node layernorm + src/dst linears ----------------------
__global__ __launch_bounds__(160)
void node_ln_lin_kernel(const float* __restrict__ node,
                        const float* __restrict__ gamma_s, const float* __restrict__ beta_s,
                        const float* __restrict__ gamma_v,
                        const float* __restrict__ Ws_src, const float* __restrict__ Ws_dst,
                        const float* __restrict__ Wv_src, const float* __restrict__ Wv_dst,
                        float* __restrict__ ss_src, float* __restrict__ ss_dst,
                        float* __restrict__ vv_src, float* __restrict__ vv_dst) {
    const int i = blockIdx.x, t = threadIdx.x;
    __shared__ float sn[64], vn[96], red[64];
    __shared__ float stat[3]; // mu, rstd, vinv
    if (t < 64) sn[t] = node[(size_t)i * NODEW + t];
    if (t >= 64) vn[t - 64] = node[(size_t)i * NODEW + t];
    __syncthreads();

    if (t < 64) red[t] = sn[t];
    __syncthreads();
    for (int off = 32; off; off >>= 1) { if (t < off) red[t] += red[t + off]; __syncthreads(); }
    if (t == 0) stat[0] = red[0] * (1.f / 64.f);
    __syncthreads();
    const float mu = stat[0];
    if (t < 64) { float dd = sn[t] - mu; red[t] = dd * dd; }
    __syncthreads();
    for (int off = 32; off; off >>= 1) { if (t < off) red[t] += red[t + off]; __syncthreads(); }
    if (t == 0) stat[1] = rsqrtf(red[0] * (1.f / 64.f) + EPS_F);
    __syncthreads();
    if (t < 64) red[t] = vn[t] * vn[t] + ((t < 32) ? vn[64 + t] * vn[64 + t] : 0.f);
    __syncthreads();
    for (int off = 32; off; off >>= 1) { if (t < off) red[t] += red[t + off]; __syncthreads(); }
    if (t == 0) stat[2] = 1.f / sqrtf(red[0] * (1.f / 32.f) + EPS_F);
    __syncthreads();

    if (t < 64) sn[t] = (sn[t] - mu) * stat[1] * gamma_s[t] + beta_s[t];
    if (t >= 64) { int j = t - 64; vn[j] = vn[j] * stat[2] * gamma_v[j / 3]; }
    __syncthreads();

    if (t < 64) {
        float a0 = 0.f, a1 = 0.f;
        for (int c = 0; c < 64; ++c) {
            float x = sn[c];
            a0 += x * Ws_src[c * 64 + t];
            a1 += x * Ws_dst[c * 64 + t];
        }
        ss_src[(size_t)i * 64 + t] = a0;
        ss_dst[(size_t)i * 64 + t] = a1;
    } else {
        int j = t - 64, d = j / 3, x = j % 3;
        float a0 = 0.f, a1 = 0.f;
        for (int c = 0; c < 32; ++c) {
            float vvv = vn[c * 3 + x];
            a0 += vvv * Wv_src[c * 32 + d];
            a1 += vvv * Wv_dst[c * 32 + d];
        }
        vv_src[(size_t)i * 96 + j] = a0;
        vv_dst[(size_t)i * 96 + j] = a1;
    }
}

// --------- edge stage 1: gather + rbf MLP + CG tensor product #1 -----------
__global__ __launch_bounds__(64)
void edge_stage1_kernel(const int* __restrict__ ei,
                        const float* __restrict__ rbf, const float* __restrict__ rsh,
                        const float* __restrict__ ss_src, const float* __restrict__ ss_dst,
                        const float* __restrict__ vv_src, const float* __restrict__ vv_dst,
                        const float* __restrict__ W_rbf,
                        _Float16* __restrict__ m1s_h, _Float16* __restrict__ m1v_h, int E_) {
    const int e = blockIdx.x, t = threadIdx.x;
    __shared__ float ms[64], mv[96], w1[224], rb[16], y[4];
    const int s = ei[e], d = ei[E_ + e];
    if (t < 16) rb[t] = rbf[(size_t)e * NBF + t];
    if (t < 4)  y[t]  = rsh[(size_t)e * 4 + t];
    ms[t] = ss_src[(size_t)s * 64 + t] + ss_dst[(size_t)d * 64 + t];
    for (int j = t; j < 96; j += 64)
        mv[j] = vv_src[(size_t)s * 96 + j] + vv_dst[(size_t)d * 96 + j];
    __syncthreads();
    for (int j = t; j < 224; j += 64) {
        float acc = 0.f;
#pragma unroll
        for (int k = 0; k < 16; ++k) acc += rb[k] * W_rbf[k * 224 + j];
        w1[j] = acc;
    }
    __syncthreads();
    const float y0 = y[0], yx = y[1], yy = y[2], yz = y[3];
    // m1_s: [w_ss*s*y0 (64) | w_vv0*dot(v,y1)/sqrt3 (32)]
    for (int c = t; c < 96; c += 64) {
        float val;
        if (c < 64) val = w1[c] * ms[c] * y0;
        else {
            int cv = c - 64;
            float dt = mv[cv * 3] * yx + mv[cv * 3 + 1] * yy + mv[cv * 3 + 2] * yz;
            val = w1[160 + cv] * dt * (1.f / SQRT3_F);
        }
        m1s_h[(size_t)e * 96 + c] = (_Float16)val;
    }
    // m1_v rows stored as (3e+x) x 128 for the component-batched GEMM
    const size_t base = (size_t)e * 3 * 128;
    for (int c = t; c < 128; c += 64) {
        float ox, oy, oz;
        if (c < 64) {
            float w = w1[64 + c] * ms[c];
            ox = w * yx; oy = w * yy; oz = w * yz;
        } else if (c < 96) {
            int cv = c - 64; float w = w1[128 + cv] * y0;
            ox = w * mv[cv * 3]; oy = w * mv[cv * 3 + 1]; oz = w * mv[cv * 3 + 2];
        } else {
            int cv = c - 96; float w = w1[192 + cv] * (1.f / SQRT2_F);
            float vx = mv[cv * 3], vy = mv[cv * 3 + 1], vz = mv[cv * 3 + 2];
            ox = w * (vy * yz - vz * yy);
            oy = w * (vz * yx - vx * yz);
            oz = w * (vx * yy - vy * yx);
        }
        m1v_h[base + c]       = (_Float16)ox;
        m1v_h[base + 128 + c] = (_Float16)oy;
        m1v_h[base + 256 + c] = (_Float16)oz;
    }
}

// ------------ attention logits: leakyrelu+dot, segment max -----------------
__global__ void edge_alpha_kernel(const _Float16* __restrict__ aval_h,
                                  const float* __restrict__ alpha_dot,
                                  const int* __restrict__ ei,
                                  float* __restrict__ aE, unsigned* __restrict__ amaxU,
                                  int E_) {
    int e = blockIdx.x * blockDim.x + threadIdx.x;
    if (e >= E_) return;
    const _Float16* ap = aval_h + (size_t)e * 128;
    const int d = ei[E_ + e];
#pragma unroll
    for (int h = 0; h < HH; ++h) {
        float acc = 0.f;
#pragma unroll
        for (int k = 0; k < ASD; ++k) {
            float x = (float)ap[h * ASD + k];
            x = x > 0.f ? x : 0.2f * x;
            acc += x * alpha_dot[h * ASD + k];
        }
        aE[(size_t)e * HH + h] = acc;
        atomicMax(&amaxU[(size_t)d * HH + h], encOrd(acc));
    }
}

// ----- edge stage 2: softmax-num/den, gating, CG tensor product #2 ---------
__global__ __launch_bounds__(64)
void edge_stage2_kernel(const int* __restrict__ ei, const float* __restrict__ rsh,
                        const _Float16* __restrict__ aval_h,
                        const _Float16* __restrict__ gatepre_h,
                        const _Float16* __restrict__ valv_h,
                        const float* __restrict__ w2_ss, const float* __restrict__ w2_sv,
                        const float* __restrict__ w2_vs, const float* __restrict__ w2_vv0,
                        const float* __restrict__ w2_vv1,
                        const float* __restrict__ aE, const unsigned* __restrict__ amaxU,
                        float* __restrict__ den, float* __restrict__ ex,
                        _Float16* __restrict__ m2s_h, _Float16* __restrict__ m2v_h, int E_) {
    const int e = blockIdx.x, t = threadIdx.x;
    __shared__ float vs[64], gate[32], vv3[96], y[4];
    const int d = ei[E_ + e];
    if (t < 4) y[t] = rsh[(size_t)e * 4 + t];
    vs[t] = sigmoidf_((float)aval_h[(size_t)e * 128 + 64 + t]);
    if (t < 32) gate[t] = sigmoidf_((float)gatepre_h[(size_t)e * 32 + t]);
    if (t < 4) {
        float a  = aE[(size_t)e * HH + t];
        float mx = decOrd(amaxU[(size_t)d * HH + t]);
        float ev = expf(a - mx);
        ex[(size_t)e * HH + t] = ev;
        atomicAdd(&den[(size_t)d * HH + t], ev);
    }
    __syncthreads();
    for (int j = t; j < 96; j += 64) {
        int c = j / 3, x = j % 3;
        vv3[j] = (float)valv_h[((size_t)e * 3 + x) * 32 + c] * gate[c];
    }
    __syncthreads();
    const float y0 = y[0], yx = y[1], yy = y[2], yz = y[3];
    for (int c = t; c < 96; c += 64) {
        float val;
        if (c < 64) val = w2_ss[c] * vs[c] * y0;
        else {
            int cv = c - 64;
            float dt = vv3[cv * 3] * yx + vv3[cv * 3 + 1] * yy + vv3[cv * 3 + 2] * yz;
            val = w2_vv0[cv] * dt * (1.f / SQRT3_F);
        }
        m2s_h[(size_t)e * 96 + c] = (_Float16)val;
    }
    const size_t base = (size_t)e * 3 * 128;
    for (int c = t; c < 128; c += 64) {
        float ox, oy, oz;
        if (c < 64) {
            float w = w2_sv[c] * vs[c];
            ox = w * yx; oy = w * yy; oz = w * yz;
        } else if (c < 96) {
            int cv = c - 64; float w = w2_vs[cv] * y0;
            ox = w * vv3[cv * 3]; oy = w * vv3[cv * 3 + 1]; oz = w * vv3[cv * 3 + 2];
        } else {
            int cv = c - 96; float w = w2_vv1[cv] * (1.f / SQRT2_F);
            float vx = vv3[cv * 3], vy = vv3[cv * 3 + 1], vz = vv3[cv * 3 + 2];
            ox = w * (vy * yz - vz * yy);
            oy = w * (vz * yx - vx * yz);
            oz = w * (vx * yy - vy * yx);
        }
        m2v_h[base + c]       = (_Float16)ox;
        m2v_h[base + 128 + c] = (_Float16)oy;
        m2v_h[base + 256 + c] = (_Float16)oz;
    }
}

// ------------------- alpha-weighted segment-sum scatter --------------------
__global__ __launch_bounds__(160)
void edge_aggregate_kernel(const int* __restrict__ ei,
                           const float* __restrict__ ex, const float* __restrict__ den,
                           const _Float16* __restrict__ hs_h, const _Float16* __restrict__ hv_h,
                           float* __restrict__ agg, int E_) {
    const int e = blockIdx.x, t = threadIdx.x;
    __shared__ float al[4];
    const int d = ei[E_ + e];
    if (t < 4) al[t] = ex[(size_t)e * HH + t] / (den[(size_t)d * HH + t] + 1e-16f);
    __syncthreads();
    const int h = t / 40, k = t % 40;
    float val;
    if (k < ASD) {
        val = (float)hs_h[(size_t)e * 64 + h * ASD + k];
    } else {
        int k2 = k - ASD, av = k2 / 3, x = k2 % 3;
        val = (float)hv_h[((size_t)e * 3 + x) * 32 + h * AVD + av];
    }
    atomicAdd(&agg[(size_t)d * 160 + t], al[h] * val);
}

// ----------------- output projection + residual ----------------------------
__global__ __launch_bounds__(160)
void node_out_kernel(const float* __restrict__ node, const float* __restrict__ agg,
                     const float* __restrict__ Wps, const float* __restrict__ Wpv,
                     float* __restrict__ out) {
    const int i = blockIdx.x, t = threadIdx.x;
    __shared__ float ag[160];
    ag[t] = agg[(size_t)i * 160 + t];
    __syncthreads();
    if (t < 64) {
        float acc = 0.f;
        for (int c = 0; c < 64; ++c)
            acc += ag[(c >> 4) * 40 + (c & 15)] * Wps[c * 64 + t];
        out[(size_t)i * NODEW + t] = node[(size_t)i * NODEW + t] + acc;
    } else {
        int j = t - 64, dd = j / 3, x = j % 3;
        float acc = 0.f;
        for (int c = 0; c < 32; ++c)
            acc += ag[(c >> 3) * 40 + 16 + (c & 7) * 3 + x] * Wpv[c * 32 + dd];
        out[(size_t)i * NODEW + t] = node[(size_t)i * NODEW + t] + acc;
    }
}

// --------------------------------- host ------------------------------------

extern "C" void kernel_launch(void* const* d_in, const int* in_sizes, int n_in,
                              void* d_out, int out_size, void* d_ws, size_t ws_size,
                              hipStream_t stream) {
    const float* node      = (const float*)d_in[0];
    const int*   ei        = (const int*)  d_in[1];
    const float* rbf       = (const float*)d_in[2];
    const float* rsh       = (const float*)d_in[3];
    const float* gamma_s   = (const float*)d_in[4];
    const float* beta_s    = (const float*)d_in[5];
    const float* gamma_v   = (const float*)d_in[6];
    const float* Ws_src    = (const float*)d_in[7];
    const float* Wv_src    = (const float*)d_in[8];
    const float* Ws_dst    = (const float*)d_in[9];
    const float* Wv_dst    = (const float*)d_in[10];
    const float* W_rbf     = (const float*)d_in[11];
    const float* W_alpha   = (const float*)d_in[12];
    const float* alpha_dot = (const float*)d_in[13];
    const float* Ws_val    = (const float*)d_in[14];
    const float* Wv_val    = (const float*)d_in[15];
    const float* Wg        = (const float*)d_in[16];
    const float* w2_ss     = (const float*)d_in[17];
    const float* w2_sv     = (const float*)d_in[18];
    const float* w2_vs     = (const float*)d_in[19];
    const float* w2_vv0    = (const float*)d_in[20];
    const float* w2_vv1    = (const float*)d_in[21];
    const float* Ws_vlin   = (const float*)d_in[22];
    const float* Wv_vlin   = (const float*)d_in[23];
    const float* Wps       = (const float*)d_in[24];
    const float* Wpv       = (const float*)d_in[25];
    float* out = (float*)d_out;

    const int N = in_sizes[0] / NODEW;
    const int E = in_sizes[1] / 2;

    // ---- carve workspace (256B-aligned sub-allocations) ----
    uintptr_t p = (uintptr_t)d_ws;
    auto carve = [&](size_t bytes) -> void* {
        uintptr_t q = (p + 255u) & ~(uintptr_t)255u;
        p = q + bytes;
        return (void*)q;
    };
    float*     ss_src   = (float*)carve((size_t)N * 64 * 4);
    float*     ss_dst   = (float*)carve((size_t)N * 64 * 4);
    float*     vv_src   = (float*)carve((size_t)N * 96 * 4);
    float*     vv_dst   = (float*)carve((size_t)N * 96 * 4);
    _Float16*  m1s_h    = (_Float16*)carve((size_t)E * 96 * 2);       // reused as m2s
    _Float16*  m1v_h    = (_Float16*)carve((size_t)E * 3 * 128 * 2);  // reused as m2v
    _Float16*  aval_h   = (_Float16*)carve((size_t)E * 128 * 2);
    _Float16*  gatepre_h= (_Float16*)carve((size_t)E * 32 * 2);
    _Float16*  valv_h   = (_Float16*)carve((size_t)E * 3 * 32 * 2);
    _Float16*  hs_h     = (_Float16*)carve((size_t)E * 64 * 2);
    _Float16*  hv_h     = (_Float16*)carve((size_t)E * 3 * 32 * 2);
    float*     aE       = (float*)carve((size_t)E * HH * 4);
    float*     ex       = (float*)carve((size_t)E * HH * 4);
    unsigned*  amaxU    = (unsigned*)carve((size_t)N * HH * 4);
    float*     den      = (float*)carve((size_t)N * HH * 4);
    float*     agg      = (float*)carve((size_t)N * 160 * 4);
    _Float16*  Bwa      = (_Float16*)carve((size_t)96 * 64 * 2);   // W_alpha
    _Float16*  Bwv      = (_Float16*)carve((size_t)96 * 64 * 2);   // Ws_val
    _Float16*  Bvv      = (_Float16*)carve((size_t)128 * 32 * 2);  // Wv_val
    _Float16*  Bg       = (_Float16*)carve((size_t)64 * 32 * 2);   // Wg
    _Float16*  Bsl      = (_Float16*)carve((size_t)96 * 64 * 2);   // Ws_vlin
    _Float16*  Bvl      = (_Float16*)carve((size_t)128 * 32 * 2);  // Wv_vlin
    if (p - (uintptr_t)d_ws > ws_size) return;  // insufficient workspace

    _Float16* m2s_h = m1s_h;   // m1s dead after its GEMMs
    _Float16* m2v_h = m1v_h;   // m1v dead after its GEMM

    const int mtilesE  = E / 16;        // 20000
    const int mtiles3E = (E * 3) / 16;  // 60000
    const int gE  = (mtilesE  + 31) / 32;   // ~4 tiles per wave
    const int g3E = (mtiles3E + 31) / 32;

    // 1) pack weights to f16
    pack_f16_kernel<<<(96 * 64 + 255) / 256, 256, 0, stream>>>(W_alpha, Bwa, 96 * 64);
    pack_f16_kernel<<<(96 * 64 + 255) / 256, 256, 0, stream>>>(Ws_val, Bwv, 96 * 64);
    pack_f16_kernel<<<(128 * 32 + 255) / 256, 256, 0, stream>>>(Wv_val, Bvv, 128 * 32);
    pack_f16_kernel<<<(64 * 32 + 255) / 256, 256, 0, stream>>>(Wg, Bg, 64 * 32);
    pack_f16_kernel<<<(96 * 64 + 255) / 256, 256, 0, stream>>>(Ws_vlin, Bsl, 96 * 64);
    pack_f16_kernel<<<(128 * 32 + 255) / 256, 256, 0, stream>>>(Wv_vlin, Bvl, 128 * 32);
    // 2) init accumulators
    init_kernel<<<1024, 256, 0, stream>>>(amaxU, den, agg, N * HH, (size_t)N * 160);
    // 3) node LN + src/dst linears
    node_ln_lin_kernel<<<N, 160, 0, stream>>>(node, gamma_s, beta_s, gamma_v,
                                              Ws_src, Ws_dst, Wv_src, Wv_dst,
                                              ss_src, ss_dst, vv_src, vv_dst);
    // 4) edge gather + rbf MLP + CG-TP #1
    edge_stage1_kernel<<<E, 64, 0, stream>>>(ei, rbf, rsh, ss_src, ss_dst,
                                             vv_src, vv_dst, W_rbf, m1s_h, m1v_h, E);
    // 5a) m1_s @ W_alpha -> aval cols [0,64)
    wmma_gemm_kernel<96, 4, 96, 128><<<gE, 256, 0, stream>>>(m1s_h, Bwa, aval_h, mtilesE);
    // 5b) m1_s @ Ws_val -> aval cols [64,128)
    wmma_gemm_kernel<96, 4, 96, 128><<<gE, 256, 0, stream>>>(m1s_h, Bwv, aval_h + 64, mtilesE);
    // 6) attention logits + segment max
    edge_alpha_kernel<<<(E + 255) / 256, 256, 0, stream>>>(aval_h, alpha_dot, ei, aE, amaxU, E);
    // 7) gate GEMM: val_s_pre @ Wg -> (E x 32)
    wmma_gemm_kernel<64, 2, 128, 32><<<gE, 256, 0, stream>>>(aval_h + 64, Bg, gatepre_h, mtilesE);
    // 8) val_v GEMM: m1_v @ Wv_val -> (3E x 32)
    wmma_gemm_kernel<128, 2, 128, 32><<<g3E, 256, 0, stream>>>(m1v_h, Bvv, valv_h, mtiles3E);
    // 9) softmax num/den, gating, CG-TP #2
    edge_stage2_kernel<<<E, 64, 0, stream>>>(ei, rsh, aval_h, gatepre_h, valv_h,
                                             w2_ss, w2_sv, w2_vs, w2_vv0, w2_vv1,
                                             aE, amaxU, den, ex, m2s_h, m2v_h, E);
    // 10) hs GEMM: m2_s @ Ws_vlin -> (E x 64)
    wmma_gemm_kernel<96, 4, 96, 64><<<gE, 256, 0, stream>>>(m2s_h, Bsl, hs_h, mtilesE);
    // 11) hv GEMM: m2_v @ Wv_vlin -> (3E x 32)
    wmma_gemm_kernel<128, 2, 128, 32><<<g3E, 256, 0, stream>>>(m2v_h, Bvl, hv_h, mtiles3E);
    // 12) alpha-weighted scatter into nodes
    edge_aggregate_kernel<<<E, 160, 0, stream>>>(ei, ex, den, hs_h, hv_h, agg, E);
    // 13) output projection + residual
    node_out_kernel<<<N, 160, 0, stream>>>(node, agg, Wps, Wpv, out);

    (void)n_in; (void)out_size;
}
